// SyllaBERT_44985487458884
// MI455X (gfx1250) — matrix operand order repaired and checked by reading, not compiled, not run on hardware
//
#include <hip/hip_runtime.h>

// ---------------------------------------------------------------------------
// Types for CDNA5 WMMA (gfx1250, wave32)
// ---------------------------------------------------------------------------
typedef __bf16 v16bf __attribute__((ext_vector_type(16)));
typedef float  v8f   __attribute__((ext_vector_type(8)));

#define WMMA_BF16(a, b, c) \
  __builtin_amdgcn_wmma_f32_16x16x32_bf16(false, (a), false, (b), (short)0, (c), false, false)

// ---------------------------------------------------------------------------
// Fragment loaders (A/B: this code's contiguous K order; C/D per CDNA5 ISA 7.12.2).
//   A 16x32 bf16 : lane -> M = lane&15 ; K = half*16 + j
//   B 32x16 bf16 : lane -> N = lane&15 ; K = half*16 + j
//   C/D 16x16 f32: lane -> N = lane&15 ; M = r + 8*half
// Main-loop loaders are UNGUARDED (rows/cols clamped by caller; K%32==0).
// ---------------------------------------------------------------------------
__device__ __forceinline__ v16bf pack16(float4 u0, float4 u1, float4 u2, float4 u3) {
  v16bf v;
  v[0] = (__bf16)u0.x;  v[1] = (__bf16)u0.y;  v[2] = (__bf16)u0.z;  v[3] = (__bf16)u0.w;
  v[4] = (__bf16)u1.x;  v[5] = (__bf16)u1.y;  v[6] = (__bf16)u1.z;  v[7] = (__bf16)u1.w;
  v[8] = (__bf16)u2.x;  v[9] = (__bf16)u2.y;  v[10] = (__bf16)u2.z; v[11] = (__bf16)u2.w;
  v[12] = (__bf16)u3.x; v[13] = (__bf16)u3.y; v[14] = (__bf16)u3.z; v[15] = (__bf16)u3.w;
  return v;
}

__device__ __forceinline__ v16bf load_a_main(const float* __restrict__ p, int k0, int half) {
  const float* q = p + k0 + half * 16;
  return pack16(*(const float4*)(q), *(const float4*)(q + 4),
                *(const float4*)(q + 8), *(const float4*)(q + 12));
}

__device__ __forceinline__ v16bf load_b_main(const float* __restrict__ p, int k0, int half) {
  const float* q = p + k0 + half * 16;
  return pack16(*(const float4*)(q), *(const float4*)(q + 4),
                *(const float4*)(q + 8), *(const float4*)(q + 12));
}

__device__ __forceinline__ v16bf load_b_strided(const float* __restrict__ p, long ldwk,
                                                int k0, int half) {
  v16bf b;
#pragma unroll
  for (int j = 0; j < 16; ++j) {
    long k = (long)(k0 + half * 16 + j);
    b[j] = (__bf16)p[k * ldwk];
  }
  return b;
}

// K-tail loaders: zero-fill beyond K (zeros required along K only).
__device__ __forceinline__ v16bf load_a_tail(const float* __restrict__ p, int k0, int K, int half) {
  v16bf a;
#pragma unroll
  for (int j = 0; j < 8; ++j) {
    int k = k0 + half * 16 + j;
    a[j] = (k < K) ? (__bf16)p[k] : (__bf16)0.0f;
    int k2 = k0 + half * 16 + 8 + j;
    a[8 + j] = (k2 < K) ? (__bf16)p[k2] : (__bf16)0.0f;
  }
  return a;
}

__device__ __forceinline__ v16bf load_b_tail(const float* __restrict__ p, long ldwk,
                                             int k0, int K, int half) {
  v16bf b;
#pragma unroll
  for (int j = 0; j < 16; ++j) {
    int k = k0 + half * 16 + j;
    b[j] = (k < K) ? (__bf16)p[(long)k * ldwk] : (__bf16)0.0f;
  }
  return b;
}

__device__ __forceinline__ void store_tile(v8f acc, float* __restrict__ C, long ldc,
                                           int tileM, int col, int M, int N,
                                           const float* __restrict__ bias,
                                           float alpha, int relu) {
  if (col >= N) return;
  const int half = (threadIdx.x >> 4) & 1;
  const float bv = bias ? bias[col] : 0.0f;
#pragma unroll
  for (int r = 0; r < 8; ++r) {
    int m = tileM + r + half * 8;
    if (m < M) {
      float v = acc[r] * alpha + bv;
      if (relu) v = fmaxf(v, 0.0f);
      C[(long)m * ldc + col] = v;
    }
  }
}

// ---------------------------------------------------------------------------
// Generic WMMA GEMM:  C[M,N] = act( alpha * A[M,K] @ W[N,K]^T + bias[N] )
// A: row stride lda, unit K stride.  W: element (n,k) at W + n*ldwn + k*ldwk.
// blockIdx.z decomposed as z = zb*zH + zh with independent strides per axis.
// Block = 128 threads (4 waves); wave tile = 32(M) x 64(N); block = 128x64.
// Out-of-range rows/cols are CLAMPED on load (their outputs are masked on
// store); only the K dimension is zero-padded (uniform tail loop).
// ---------------------------------------------------------------------------
__global__ void __launch_bounds__(128)
gemm_wmma_kernel(const float* __restrict__ A, long lda,
                 const float* __restrict__ W, long ldwn, long ldwk,
                 const float* __restrict__ bias,
                 float* __restrict__ C, long ldc,
                 int M, int N, int K, float alpha, int relu, int zH,
                 long aSB, long aSH, long wSB, long wSH, long cSB, long cSH) {
  const int z = blockIdx.z;
  const int zb = z / zH;
  const int zh = z - zb * zH;
  A += (long)zb * aSB + (long)zh * aSH;
  W += (long)zb * wSB + (long)zh * wSH;
  C += (long)zb * cSB + (long)zh * cSH;

  const int wave = threadIdx.x >> 5;
  const int lane = threadIdx.x & 31;
  const int l16 = lane & 15;
  const int half = lane >> 4;
  const int tileM = blockIdx.x * 128 + wave * 32;
  const int tileN = blockIdx.y * 64;
  if (tileM >= M) return;

  // Clamped row/col base pointers (branch-free hot loop).
  int row0 = tileM + l16;        row0 = (row0 < M) ? row0 : (M - 1);
  int row1 = tileM + 16 + l16;   row1 = (row1 < M) ? row1 : (M - 1);
  const float* ap0 = A + (long)row0 * lda;
  const float* ap1 = A + (long)row1 * lda;
  int c0 = tileN + l16;       c0 = (c0 < N) ? c0 : (N - 1);
  int c1 = tileN + 16 + l16;  c1 = (c1 < N) ? c1 : (N - 1);
  int c2 = tileN + 32 + l16;  c2 = (c2 < N) ? c2 : (N - 1);
  int c3 = tileN + 48 + l16;  c3 = (c3 < N) ? c3 : (N - 1);
  const float* wp0 = W + (long)c0 * ldwn;
  const float* wp1 = W + (long)c1 * ldwn;
  const float* wp2 = W + (long)c2 * ldwn;
  const float* wp3 = W + (long)c3 * ldwn;

  v8f acc0 = {}; v8f acc1 = {}; v8f acc2 = {}; v8f acc3 = {};
  v8f acc4 = {}; v8f acc5 = {}; v8f acc6 = {}; v8f acc7 = {};

  const int Kmain = K & ~31;
  if (ldwk == 1) {
    for (int k0 = 0; k0 < Kmain; k0 += 32) {
      v16bf a0 = load_a_main(ap0, k0, half);
      v16bf a1 = load_a_main(ap1, k0, half);
      v16bf b0 = load_b_main(wp0, k0, half);
      v16bf b1 = load_b_main(wp1, k0, half);
      v16bf b2 = load_b_main(wp2, k0, half);
      v16bf b3 = load_b_main(wp3, k0, half);
      acc0 = WMMA_BF16(a0, b0, acc0);
      acc1 = WMMA_BF16(a0, b1, acc1);
      acc2 = WMMA_BF16(a0, b2, acc2);
      acc3 = WMMA_BF16(a0, b3, acc3);
      acc4 = WMMA_BF16(a1, b0, acc4);
      acc5 = WMMA_BF16(a1, b1, acc5);
      acc6 = WMMA_BF16(a1, b2, acc6);
      acc7 = WMMA_BF16(a1, b3, acc7);
    }
  } else {
    for (int k0 = 0; k0 < Kmain; k0 += 32) {
      v16bf a0 = load_a_main(ap0, k0, half);
      v16bf a1 = load_a_main(ap1, k0, half);
      v16bf b0 = load_b_strided(wp0, ldwk, k0, half);
      v16bf b1 = load_b_strided(wp1, ldwk, k0, half);
      v16bf b2 = load_b_strided(wp2, ldwk, k0, half);
      v16bf b3 = load_b_strided(wp3, ldwk, k0, half);
      acc0 = WMMA_BF16(a0, b0, acc0);
      acc1 = WMMA_BF16(a0, b1, acc1);
      acc2 = WMMA_BF16(a0, b2, acc2);
      acc3 = WMMA_BF16(a0, b3, acc3);
      acc4 = WMMA_BF16(a1, b0, acc4);
      acc5 = WMMA_BF16(a1, b1, acc5);
      acc6 = WMMA_BF16(a1, b2, acc6);
      acc7 = WMMA_BF16(a1, b3, acc7);
    }
  }
  if (Kmain < K) {
    v16bf a0 = load_a_tail(ap0, Kmain, K, half);
    v16bf a1 = load_a_tail(ap1, Kmain, K, half);
    v16bf b0 = load_b_tail(wp0, ldwk, Kmain, K, half);
    v16bf b1 = load_b_tail(wp1, ldwk, Kmain, K, half);
    v16bf b2 = load_b_tail(wp2, ldwk, Kmain, K, half);
    v16bf b3 = load_b_tail(wp3, ldwk, Kmain, K, half);
    acc0 = WMMA_BF16(a0, b0, acc0);
    acc1 = WMMA_BF16(a0, b1, acc1);
    acc2 = WMMA_BF16(a0, b2, acc2);
    acc3 = WMMA_BF16(a0, b3, acc3);
    acc4 = WMMA_BF16(a1, b0, acc4);
    acc5 = WMMA_BF16(a1, b1, acc5);
    acc6 = WMMA_BF16(a1, b2, acc6);
    acc7 = WMMA_BF16(a1, b3, acc7);
  }

  store_tile(acc0, C, ldc, tileM, tileN + l16, M, N, bias, alpha, relu);
  store_tile(acc1, C, ldc, tileM, tileN + 16 + l16, M, N, bias, alpha, relu);
  store_tile(acc2, C, ldc, tileM, tileN + 32 + l16, M, N, bias, alpha, relu);
  store_tile(acc3, C, ldc, tileM, tileN + 48 + l16, M, N, bias, alpha, relu);
  store_tile(acc4, C, ldc, tileM + 16, tileN + l16, M, N, bias, alpha, relu);
  store_tile(acc5, C, ldc, tileM + 16, tileN + 16 + l16, M, N, bias, alpha, relu);
  store_tile(acc6, C, ldc, tileM + 16, tileN + 32 + l16, M, N, bias, alpha, relu);
  store_tile(acc7, C, ldc, tileM + 16, tileN + 48 + l16, M, N, bias, alpha, relu);
}

// ---------------------------------------------------------------------------
// conv0: (B,1,160000) -> (B, 31999, 512), kernel 10 stride 5, ReLU, NTC layout
// ---------------------------------------------------------------------------
__global__ void conv0_kernel(const float* __restrict__ x, const float* __restrict__ w,
                             const float* __restrict__ b, float* __restrict__ y,
                             int Tin, int Tout, int Co, int Kc, int Stc) {
  long idx = (long)blockIdx.x * blockDim.x + threadIdx.x;
  long total = (long)Tout * Co;
  if (idx >= total) return;
  int batch = blockIdx.y;
  int t = (int)(idx / Co);
  int co = (int)(idx - (long)t * Co);
  const float* xp = x + (long)batch * Tin + (long)t * Stc;
  const float* wp = w + (long)co * Kc;
  float s = b[co];
  for (int kk = 0; kk < Kc; ++kk) s = fmaf(xp[kk], wp[kk], s);
  y[((long)batch * Tout + t) * Co + co] = fmaxf(s, 0.0f);
}

// Repack conv weight (Co,Ci,Kc) -> (Co, Kc*Ci) so im2col rows are contiguous.
__global__ void repack_w_kernel(const float* __restrict__ w, float* __restrict__ wr,
                                int Co, int Ci, int Kc) {
  long idx = (long)blockIdx.x * blockDim.x + threadIdx.x;
  long total = (long)Co * Ci * Kc;
  if (idx >= total) return;
  int co = (int)(idx / ((long)Ci * Kc));
  long rem = idx - (long)co * Ci * Kc;
  int kk = (int)(rem / Ci);
  int ci = (int)(rem - (long)kk * Ci);
  wr[(long)co * Kc * Ci + (long)kk * Ci + ci] = w[(long)co * Ci * Kc + (long)ci * Kc + kk];
}

__global__ void add_pos_kernel(float* __restrict__ x, const float* __restrict__ pos,
                               long total, int T, int E) {
  long i = (long)blockIdx.x * blockDim.x + threadIdx.x;
  if (i >= total) return;
  long bt = i / E;
  int e = (int)(i - bt * E);
  int t = (int)(bt % T);
  x[i] += pos[(long)t * E + e];
}

// Row softmax (length n, row stride ld), one wave per row.
__global__ void softmax_rows_kernel(float* __restrict__ S, int rows, int n, int ld) {
  int row = blockIdx.x * (blockDim.x >> 5) + (threadIdx.x >> 5);
  if (row >= rows) return;
  int lane = threadIdx.x & 31;
  float* p = S + (long)row * ld;
  float mx = -3.4e38f;
  for (int j = lane; j < n; j += 32) mx = fmaxf(mx, p[j]);
  for (int o = 16; o; o >>= 1) mx = fmaxf(mx, __shfl_xor(mx, o, 32));
  float sum = 0.0f;
  for (int j = lane; j < n; j += 32) { float e = __expf(p[j] - mx); p[j] = e; sum += e; }
  for (int o = 16; o; o >>= 1) sum += __shfl_xor(sum, o, 32);
  float inv = 1.0f / sum;
  for (int j = lane; j < n; j += 32) p[j] *= inv;
}

// y = LayerNorm(x + delta) * g + b  (delta may be null). One wave per row, E<=1024.
__global__ void add_ln_kernel(const float* __restrict__ x, const float* __restrict__ dlt,
                              const float* __restrict__ g, const float* __restrict__ bta,
                              float* __restrict__ y, int rows, int E) {
  int row = blockIdx.x * (blockDim.x >> 5) + (threadIdx.x >> 5);
  if (row >= rows) return;
  int lane = threadIdx.x & 31;
  const float* xp = x + (long)row * E;
  const float* dp = dlt ? dlt + (long)row * E : nullptr;
  float* yp = y + (long)row * E;
  float vals[32];
  int cnt = 0;
  float s = 0.0f;
  for (int j = lane; j < E; j += 32) {
    float v = xp[j] + (dp ? dp[j] : 0.0f);
    vals[cnt++] = v;
    s += v;
  }
  for (int o = 16; o; o >>= 1) s += __shfl_xor(s, o, 32);
  float mean = s / (float)E;
  float vs = 0.0f;
  for (int i = 0; i < cnt; ++i) { float d = vals[i] - mean; vs += d * d; }
  for (int o = 16; o; o >>= 1) vs += __shfl_xor(vs, o, 32);
  float rstd = rsqrtf(vs / (float)E + 1e-5f);
  cnt = 0;
  for (int j = lane; j < E; j += 32) {
    yp[j] = (vals[cnt++] - mean) * rstd * g[j] + bta[j];
  }
}

__global__ void zero_kernel(float* __restrict__ p, long n) {
  long i = (long)blockIdx.x * blockDim.x + threadIdx.x;
  if (i < n) p[i] = 0.0f;
}

__global__ void pool_scatter_kernel(const float* __restrict__ x, const int* __restrict__ seg,
                                    float* __restrict__ sums, float* __restrict__ cnt,
                                    int T, int E, int S, long total) {
  long i = (long)blockIdx.x * blockDim.x + threadIdx.x;
  if (i >= total) return;
  long bt = i / E;
  int e = (int)(i - bt * E);
  int b = (int)(bt / T);
  int g = seg[bt] + b * S;
  atomicAdd(&sums[(long)g * E + e], x[i]);
  if (e == 0) atomicAdd(&cnt[g], 1.0f);
}

__global__ void pool_div_kernel(float* __restrict__ sums, const float* __restrict__ cnt,
                                int E, long n) {
  long i = (long)blockIdx.x * blockDim.x + threadIdx.x;
  if (i >= n) return;
  long g = i / E;
  sums[i] = sums[i] / fmaxf(cnt[g], 1.0f);
}

// ---------------------------------------------------------------------------
// Host side
// ---------------------------------------------------------------------------
static void launch_gemm(hipStream_t stream, const float* A, long lda,
                        const float* W, long ldwn, long ldwk, const float* bias,
                        float* C, long ldc, int M, int N, int K, float alpha, int relu,
                        int gz, int zH, long aSB, long aSH, long wSB, long wSH,
                        long cSB, long cSH) {
  dim3 grid((unsigned)((M + 127) / 128), (unsigned)((N + 63) / 64), (unsigned)gz);
  gemm_wmma_kernel<<<grid, dim3(128), 0, stream>>>(A, lda, W, ldwn, ldwk, bias, C, ldc,
                                                   M, N, K, alpha, relu, zH,
                                                   aSB, aSH, wSB, wSH, cSB, cSH);
}

extern "C" void kernel_launch(void* const* d_in, const int* in_sizes, int n_in,
                              void* d_out, int out_size, void* d_ws, size_t ws_size,
                              hipStream_t stream) {
  (void)in_sizes; (void)n_in; (void)out_size; (void)ws_size;

  constexpr int Bn = 8, E = 768, Hh = 12, Ll = 12, FFd = 2048, Sseg = 50, NC = 100;
  constexpr int Tt = 499, BT = Bn * Tt, Tp = 512, DH = 64;
  const int Kk[7] = {10, 3, 3, 3, 2, 2, 2};
  const int St[7] = {5, 2, 2, 2, 2, 2, 2};

  const float* x_raw = (const float*)d_in[0];
  const int*   seg   = (const int*)d_in[1];
  const float* cw[7]; const float* cb[7];
  for (int i = 0; i < 7; ++i) { cw[i] = (const float*)d_in[2 + 2 * i]; cb[i] = (const float*)d_in[3 + 2 * i]; }
  const float* pos    = (const float*)d_in[16];
  const float* qkv_w  = (const float*)d_in[17];
  const float* qkv_b  = (const float*)d_in[18];
  const float* out_w  = (const float*)d_in[19];
  const float* out_b  = (const float*)d_in[20];
  const float* ff1_w  = (const float*)d_in[21];
  const float* ff1_b  = (const float*)d_in[22];
  const float* ff2_w  = (const float*)d_in[23];
  const float* ff2_b  = (const float*)d_in[24];
  const float* ln1_w  = (const float*)d_in[25];
  const float* ln1_b  = (const float*)d_in[26];
  const float* ln2_w  = (const float*)d_in[27];
  const float* ln2_b  = (const float*)d_in[28];
  const float* lnf_w  = (const float*)d_in[29];
  const float* lnf_b  = (const float*)d_in[30];
  const float* proj_w = (const float*)d_in[31];
  const float* proj_b = (const float*)d_in[32];

  float* ws = (float*)d_ws;
  size_t off = 0;
  auto wsalloc = [&](size_t n) -> float* {
    float* p = ws + off;
    off += (n + 15) & ~(size_t)15;
    return p;
  };
  float* bufEven = wsalloc((size_t)Bn * 31999 * 512);
  float* bufOdd  = wsalloc((size_t)Bn * 15999 * 512);
  float* wrep    = wsalloc((size_t)786432);
  float* x0      = wsalloc((size_t)BT * E);
  float* x1      = wsalloc((size_t)BT * E);
  float* qkv     = wsalloc((size_t)BT * 3 * E);
  float* Sb      = wsalloc((size_t)Bn * Hh * Tt * Tp);
  float* ao      = wsalloc((size_t)BT * E);
  float* ffh     = wsalloc((size_t)BT * FFd);
  float* tmp     = wsalloc((size_t)BT * E);
  float* psum    = wsalloc((size_t)Bn * Sseg * E);
  float* pcnt    = wsalloc((size_t)Bn * Sseg);

  // ---- Feature extractor ----
  {
    long total = (long)31999 * 512;
    dim3 grid((unsigned)((total + 255) / 256), Bn);
    conv0_kernel<<<grid, 256, 0, stream>>>(x_raw, cw[0], cb[0], bufEven, 160000, 31999, 512, Kk[0], St[0]);
  }
  float* convin = bufEven;
  float* convout = bufOdd;
  int Tcur = 31999;
  for (int i = 1; i <= 6; ++i) {
    int Ci = 512, Co = (i == 6) ? 768 : 512, Kc = Kk[i], Stc = St[i];
    int Tout = (Tcur - Kc) / Stc + 1;
    long wn = (long)Co * Ci * Kc;
    repack_w_kernel<<<dim3((unsigned)((wn + 255) / 256)), 256, 0, stream>>>(cw[i], wrep, Co, Ci, Kc);
    int K = Kc * Ci;
    float* outp = (i == 6) ? x0 : convout;
    launch_gemm(stream, convin, (long)Stc * Ci, wrep, (long)K, 1, cb[i], outp, (long)Co,
                Tout, Co, K, 1.0f, /*relu=*/1, /*gz=*/Bn, /*zH=*/1,
                (long)Tcur * Ci, 0, 0, 0, (long)Tout * Co, 0);
    convout = convin;
    convin = outp;
    Tcur = Tout;
  }

  // ---- Positional embedding ----
  {
    long total = (long)BT * E;
    add_pos_kernel<<<dim3((unsigned)((total + 255) / 256)), 256, 0, stream>>>(x0, pos, total, Tt, E);
  }

  // ---- Transformer encoder (post-LN) ----
  const float scale = 0.125f;  // 1/sqrt(64)
  for (int l = 0; l < Ll; ++l) {
    // qkv = x0 @ qkv_w^T + qkv_b   (BT x 2304)
    launch_gemm(stream, x0, E, qkv_w + (long)l * 3 * E * E, E, 1, qkv_b + (long)l * 3 * E,
                qkv, 3 * E, BT, 3 * E, E, 1.0f, 0, 1, 1, 0, 0, 0, 0, 0, 0);
    // S = scale * Q @ K^T per (b,h)
    launch_gemm(stream, qkv, 3 * E, qkv + E, 3 * E, 1, nullptr, Sb, Tp,
                Tt, Tt, DH, scale, 0, Bn * Hh, Hh,
                (long)Tt * 3 * E, DH, (long)Tt * 3 * E, DH,
                (long)Hh * Tt * Tp, (long)Tt * Tp);
    softmax_rows_kernel<<<dim3((unsigned)((Bn * Hh * Tt + 3) / 4)), 128, 0, stream>>>(
        Sb, Bn * Hh * Tt, Tt, Tp);
    // O = P @ V per (b,h)  (B strided gather along K)
    launch_gemm(stream, Sb, Tp, qkv + 2 * E, 1, 3 * E, nullptr, ao, E,
                Tt, DH, Tt, 1.0f, 0, Bn * Hh, Hh,
                (long)Hh * Tt * Tp, (long)Tt * Tp, (long)Tt * 3 * E, DH,
                (long)Tt * E, DH);
    // out projection
    launch_gemm(stream, ao, E, out_w + (long)l * E * E, E, 1, out_b + (long)l * E,
                tmp, E, BT, E, E, 1.0f, 0, 1, 1, 0, 0, 0, 0, 0, 0);
    add_ln_kernel<<<dim3((unsigned)((BT + 3) / 4)), 128, 0, stream>>>(
        x0, tmp, ln1_w + (long)l * E, ln1_b + (long)l * E, x1, BT, E);
    // FFN
    launch_gemm(stream, x1, E, ff1_w + (long)l * FFd * E, E, 1, ff1_b + (long)l * FFd,
                ffh, FFd, BT, FFd, E, 1.0f, /*relu=*/1, 1, 1, 0, 0, 0, 0, 0, 0);
    launch_gemm(stream, ffh, FFd, ff2_w + (long)l * E * FFd, FFd, 1, ff2_b + (long)l * E,
                tmp, E, BT, E, FFd, 1.0f, 0, 1, 1, 0, 0, 0, 0, 0, 0);
    add_ln_kernel<<<dim3((unsigned)((BT + 3) / 4)), 128, 0, stream>>>(
        x1, tmp, ln2_w + (long)l * E, ln2_b + (long)l * E, x0, BT, E);
  }

  // ---- Final LN ----
  add_ln_kernel<<<dim3((unsigned)((BT + 3) / 4)), 128, 0, stream>>>(
      x0, nullptr, lnf_w, lnf_b, x1, BT, E);

  // ---- Segment mean pooling ----
  {
    long zn = (long)Bn * Sseg * E + (long)Bn * Sseg;  // psum then pcnt (contiguous)
    zero_kernel<<<dim3((unsigned)((zn + 255) / 256)), 256, 0, stream>>>(psum, zn);
    long total = (long)BT * E;
    pool_scatter_kernel<<<dim3((unsigned)((total + 255) / 256)), 256, 0, stream>>>(
        x1, seg, psum, pcnt, Tt, E, Sseg, total);
    long pn = (long)Bn * Sseg * E;
    pool_div_kernel<<<dim3((unsigned)((pn + 255) / 256)), 256, 0, stream>>>(psum, pcnt, E, pn);
  }

  // ---- Class projection -> d_out (400 x 100 f32) ----
  launch_gemm(stream, psum, E, proj_w, E, 1, proj_b, (float*)d_out, NC,
              Bn * Sseg, NC, E, 1.0f, 0, 1, 1, 0, 0, 0, 0, 0, 0);
}
